// MHSA_80461917323387
// MI455X (gfx1250) — compile-verified, matrix operand order checked by reading, not run on hardware
//
#include <hip/hip_runtime.h>
#include <hip/hip_bf16.h>

// ---------------------------------------------------------------------------
// MHSA (reference's raw-reshape semantics) for MI455X / gfx1250.
//   B=4, T=1024, D=1024, H=16, Dh=64.
// Per (b,h): heads own CONTIGUOUS 64-row slabs of Q/K/V (raw reshape!):
//   S  = temp_h * Kh^T Vh          (1024x1024, rank-64 contraction)
//   P  = softmax_rows(S)
//   Oh = Qh @ P                    (64x1024)
// bf16 WMMA (v_wmma_f32_16x16x32_bf16) with f32 accum; flash-style two-pass
// softmax so S (256 MB f32) never touches HBM. K/V are stored TRANSPOSED per
// slab by the projection kernel so all K/V tile staging is a contiguous copy
// -> GLOBAL_LOAD_ASYNC_TO_LDS_B128 (ASYNCcnt) when the toolchain exposes it.
// ---------------------------------------------------------------------------

typedef __bf16 bf16_t;
typedef __attribute__((ext_vector_type(16))) __bf16 v16bf;
typedef __attribute__((ext_vector_type(8)))  float  v8f;
typedef __attribute__((ext_vector_type(4)))  unsigned int v4u;

__device__ __forceinline__ v8f wmma_bf16(v16bf a, v16bf b, v8f c) {
  // 8 args: (neg_a, A, neg_b, B, c_mod, C, reuse_a, reuse_b)
  return __builtin_amdgcn_wmma_f32_16x16x32_bf16(false, a, false, b,
                                                 (short)0, c, false, false);
}

// ---- async global->LDS copy (16B per lane), guarded for toolchain drift ----
#if defined(__gfx1250__) && __has_builtin(__builtin_amdgcn_global_load_async_to_lds_b128)
#define USE_ASYNC_COPY 1
#else
#define USE_ASYNC_COPY 0
#endif

// Exact parameter types per the compiler's own diagnostic:
//   param0: __attribute__((vector_size(16))) int __device__(AS1) *
//   param1: same pointee in LDS (AS3)
typedef int v4i_vs __attribute__((vector_size(4 * sizeof(int))));
typedef __attribute__((address_space(1))) v4i_vs* as1_v4i;
typedef __attribute__((address_space(3))) v4i_vs* as3_v4i;

__device__ __forceinline__ void cp_b128_g2lds(const bf16_t* g, bf16_t* l) {
#if USE_ASYNC_COPY
  __builtin_amdgcn_global_load_async_to_lds_b128((as1_v4i)g, (as3_v4i)l, 0, 0);
#else
  *(v4u*)l = *(const v4u*)g;   // fallback: copy through VGPRs
#endif
}

__device__ __forceinline__ void cp_commit_wait() {
#if USE_ASYNC_COPY
#if __has_builtin(__builtin_amdgcn_s_wait_asynccnt)
  __builtin_amdgcn_s_wait_asynccnt(0);
#else
  asm volatile("s_wait_asynccnt 0x0" ::: "memory");
#endif
#endif
}

// ---------------------------------------------------------------------------
// Kernel 1: Q/K/V projection  Y = x @ W + bias  (4096x1024 @ 1024x1024).
// grid = (32 Mtiles, 8 Ntiles, 3 matrices), block = 256 (8 waves).
// Q written row-major (token, feature).  K/V written TRANSPOSED per slab:
// layout (b, h, feature, dd) so attention staging is contiguous.
// ---------------------------------------------------------------------------
__global__ __launch_bounds__(256) void qkv_proj(
    const float* __restrict__ x,
    const float* __restrict__ Wq, const float* __restrict__ bq,
    const float* __restrict__ Wk, const float* __restrict__ bk,
    const float* __restrict__ Wv, const float* __restrict__ bv,
    bf16_t* __restrict__ Qb, bf16_t* __restrict__ KbT, bf16_t* __restrict__ VbT)
{
  __shared__ bf16_t As[128][80];   // [m][k]   (x tile, bf16)
  __shared__ bf16_t Bs[128][80];   // [n][k]   (W^T tile, bf16)

  const int which = blockIdx.z;
  const float* W    = (which == 0) ? Wq : (which == 1) ? Wk : Wv;
  const float* bias = (which == 0) ? bq : (which == 1) ? bk : bv;
  bf16_t*      out  = (which == 0) ? Qb : (which == 1) ? KbT : VbT;

  const int m0 = blockIdx.x * 128;
  const int n0 = blockIdx.y * 128;
  const int tid = threadIdx.x, lane = tid & 31, wave = tid >> 5;
  const int wm = wave >> 1;          // 0..3 -> 32-row band
  const int wn = wave & 1;           // 0..1 -> 64-col band
  const int lr = lane & 15;          // intra-tile row/col
  const int lh = lane >> 4;          // lane half

  v8f zero = {};
  v8f acc[2][4];
  for (int i = 0; i < 2; ++i)
    for (int j = 0; j < 4; ++j) acc[i][j] = zero;

  for (int kk = 0; kk < 1024; kk += 64) {
    __syncthreads();
    for (int i = tid; i < 128 * 64; i += 256) {       // x tile -> As[m][k]
      int r = i >> 6, c = i & 63;
      As[r][c] = (bf16_t)x[(size_t)(m0 + r) * 1024 + kk + c];
    }
    for (int i = tid; i < 64 * 128; i += 256) {       // W tile -> Bs[n][k]
      int k = i >> 7, n = i & 127;
      Bs[n][k] = (bf16_t)W[(size_t)(kk + k) * 1024 + n0 + n];
    }
    __syncthreads();

    for (int ks = 0; ks < 64; ks += 32) {
      v16bf afr[2], bfr[4];
      for (int i = 0; i < 2; ++i)
        afr[i] = *(const v16bf*)&As[wm * 32 + i * 16 + lr][ks + lh * 16];
      for (int j = 0; j < 4; ++j)
        bfr[j] = *(const v16bf*)&Bs[wn * 64 + j * 16 + lr][ks + lh * 16];
      for (int i = 0; i < 2; ++i)
        for (int j = 0; j < 4; ++j)
          acc[i][j] = wmma_bf16(afr[i], bfr[j], acc[i][j]);
    }
  }

  // epilogue: + bias, convert, store (Q row-major; K/V slab-transposed)
  for (int i = 0; i < 2; ++i)
    for (int j = 0; j < 4; ++j) {
      int col = n0 + wn * 64 + j * 16 + lr;           // feature index
      float bc = bias[col];
      for (int e = 0; e < 8; ++e) {
        int row = m0 + wm * 32 + i * 16 + lh * 8 + e; // global token row
        bf16_t v = (bf16_t)(acc[i][j][e] + bc);
        if (which == 0) {
          out[(size_t)row * 1024 + col] = v;
        } else {
          int bb = row >> 10, tok = row & 1023;
          int hh = tok >> 6, dd = tok & 63;
          // (b, h, feature, dd)
          out[(((size_t)(bb * 16 + hh) * 1024) + col) * 64 + dd] = v;
        }
      }
    }
}

// ---------------------------------------------------------------------------
// Kernel 2: softmax statistics. For each (b,h): m[t]=rowmax(S), rz[t]=1/rowsum.
// S tiles streamed through WMMA; online (flash) max/sum in registers;
// cross-lane merge via shfl_xor inside each 16-lane half.
// grid = 64 (b*h), block = 256 (8 waves).
// ---------------------------------------------------------------------------
__global__ __launch_bounds__(256) void attn_stats(
    const bf16_t* __restrict__ KbT, const bf16_t* __restrict__ VbT,
    const float* __restrict__ temperature,
    float* __restrict__ mrow, float* __restrict__ rzrow)
{
  __shared__ bf16_t At[128][80];    // Kh^T tile: [t][dd]
  __shared__ bf16_t VsT[256][80];   // Vh^T chunk: [s][dd]

  const int bh = blockIdx.x;            // 0..63
  const int h = bh & 15;
  const size_t slab = (size_t)bh * 1024 * 64;
  const bf16_t* KhT = KbT + slab;       // [t][dd], rows of 64 contiguous
  const bf16_t* VhT = VbT + slab;       // [s][dd]
  const float tmp = temperature[h];

  const int tid = threadIdx.x, lane = tid & 31, wave = tid >> 5;
  const int lr = lane & 15, lh = lane >> 4;
  v8f zero = {};

  for (int tb = 0; tb < 8; ++tb) {
    __syncthreads();
    // Kh^T tile rows [tb*128, +128): contiguous 128B/row -> async b128 x8
    for (int i = tid; i < 128 * 8; i += 256) {
      int r = i >> 3, c = (i & 7) * 8;
      cp_b128_g2lds(&KhT[(size_t)(tb * 128 + r) * 64 + c], &At[r][c]);
    }

    float m_r[8], z_r[8];
    for (int e = 0; e < 8; ++e) { m_r[e] = -3.0e38f; z_r[e] = 0.0f; }

    for (int sc = 0; sc < 4; ++sc) {
      __syncthreads();
      for (int i = tid; i < 256 * 8; i += 256) {
        int r = i >> 3, c = (i & 7) * 8;
        cp_b128_g2lds(&VhT[(size_t)(sc * 256 + r) * 64 + c], &VsT[r][c]);
      }
      cp_commit_wait();
      __syncthreads();
      for (int st = 0; st < 16; ++st) {
        v8f c = zero;
        for (int ks = 0; ks < 64; ks += 32) {
          v16bf a  = *(const v16bf*)&At[wave * 16 + lr][ks + lh * 16];
          v16bf bb = *(const v16bf*)&VsT[st * 16 + lr][ks + lh * 16];
          c = wmma_bf16(a, bb, c);
        }
        for (int e = 0; e < 8; ++e) {
          float xv = c[e] * tmp;
          float nm = fmaxf(m_r[e], xv);
          z_r[e] = z_r[e] * __expf(m_r[e] - nm) + __expf(xv - nm);
          m_r[e] = nm;
        }
      }
    }
    // merge the 16 lane-columns of each C row (halves hold disjoint rows)
    for (int off = 1; off < 16; off <<= 1) {
      for (int e = 0; e < 8; ++e) {
        float om = __shfl_xor(m_r[e], off, 32);
        float oz = __shfl_xor(z_r[e], off, 32);
        float nm = fmaxf(m_r[e], om);
        z_r[e] = z_r[e] * __expf(m_r[e] - nm) + oz * __expf(om - nm);
        m_r[e] = nm;
      }
    }
    if (lr == 0) {
      for (int e = 0; e < 8; ++e) {
        int t = tb * 128 + wave * 16 + lh * 8 + e;
        mrow[(size_t)bh * 1024 + t]  = m_r[e];
        rzrow[(size_t)bh * 1024 + t] = 1.0f / z_r[e];
      }
    }
  }
}

// ---------------------------------------------------------------------------
// Kernel 3: O_h = Qh @ (exp(S - m) * rz).  S recomputed in tiles (rank-64!),
// normalized probabilities staged as bf16 in LDS ([s][t] so WMMA B-fragment
// reads are contiguous), then WMMA-accumulated against Q.
// grid = (64 b*h, 8 s-blocks of 128), block = 256 (8 waves).
// ---------------------------------------------------------------------------
__global__ __launch_bounds__(256) void attn_out(
    const bf16_t* __restrict__ Qb, const bf16_t* __restrict__ KbT,
    const bf16_t* __restrict__ VbT, const float* __restrict__ temperature,
    const float* __restrict__ mrow, const float* __restrict__ rzrow,
    float* __restrict__ out)
{
  __shared__ bf16_t VsT[128][80];   // [s][dd]
  __shared__ bf16_t At[64][80];     // [t][dd]
  __shared__ bf16_t Et[128][80];    // probs, [s][t]

  const int bh = blockIdx.x, sb = blockIdx.y;
  const int b = bh >> 4, h = bh & 15;
  const int s0 = sb * 128;
  const size_t slab = (size_t)bh * 1024 * 64;
  const bf16_t* Qh  = Qb + ((size_t)b * 1024 + h * 64) * 1024;  // [dd][t]
  const bf16_t* KhT = KbT + slab;                               // [t][dd]
  const bf16_t* VhT = VbT + slab;                               // [s][dd]
  const float tmp = temperature[h];

  const int tid = threadIdx.x, lane = tid & 31, wave = tid >> 5;
  const int lr = lane & 15, lh = lane >> 4;
  const int wm = wave >> 1;   // 0..3
  const int wn = wave & 1;    // 0..1

  v8f zero = {};
  v8f acc[4];
  for (int j = 0; j < 4; ++j) acc[j] = zero;

  // Vh^T block [s0, s0+128): contiguous rows -> async copy
  for (int i = tid; i < 128 * 8; i += 256) {
    int r = i >> 3, c = (i & 7) * 8;
    cp_b128_g2lds(&VhT[(size_t)(s0 + r) * 64 + c], &VsT[r][c]);
  }

  for (int tblk = 0; tblk < 16; ++tblk) {
    const int t0 = tblk * 64;
    __syncthreads();   // protect At/Et reuse
    for (int i = tid; i < 64 * 8; i += 256) {
      int r = i >> 3, c = (i & 7) * 8;
      cp_b128_g2lds(&KhT[(size_t)(t0 + r) * 64 + c], &At[r][c]);
    }
    cp_commit_wait();
    __syncthreads();

    // per-row softmax stats for this wave's 16 t-rows
    float mr[8], rzr[8];
    for (int e = 0; e < 8; ++e) {
      int t = t0 + wm * 16 + lh * 8 + e;
      mr[e]  = mrow[(size_t)bh * 1024 + t];
      rzr[e] = rzrow[(size_t)bh * 1024 + t];
    }

    // S tiles -> normalized probs into Et[s][t]
    for (int j = 0; j < 4; ++j) {
      v8f c = zero;
      for (int ks = 0; ks < 64; ks += 32) {
        v16bf a  = *(const v16bf*)&At[wm * 16 + lr][ks + lh * 16];
        v16bf bb = *(const v16bf*)&VsT[wn * 64 + j * 16 + lr][ks + lh * 16];
        c = wmma_bf16(a, bb, c);
      }
      int sl = wn * 64 + j * 16 + lr;
      for (int e = 0; e < 8; ++e) {
        int tl = wm * 16 + lh * 8 + e;
        Et[sl][tl] = (bf16_t)(__expf(c[e] * tmp - mr[e]) * rzr[e]);
      }
    }
    __syncthreads();

    // O += Qh[:, t-block] @ Et^T ; Q A-fragments straight from global
    v16bf aq[2];
    for (int ks = 0; ks < 64; ks += 32) {
      int dd = wm * 16 + lr;
      aq[ks >> 5] = *(const v16bf*)&Qh[(size_t)dd * 1024 + t0 + ks + lh * 16];
    }
    for (int j = 0; j < 4; ++j)
      for (int ks = 0; ks < 64; ks += 32) {
        v16bf bb = *(const v16bf*)&Et[wn * 64 + j * 16 + lr][ks + lh * 16];
        acc[j] = wmma_bf16(aq[ks >> 5], bb, acc[j]);
      }
  }

  // epilogue: out[b, h*64+dd, s0+scol] (raw reshape back => same slab layout)
  for (int j = 0; j < 4; ++j) {
    int scol = s0 + wn * 64 + j * 16 + lr;
    for (int e = 0; e < 8; ++e) {
      int dd = wm * 16 + lh * 8 + e;
      out[((size_t)b * 1024 + h * 64 + dd) * 1024 + scol] = acc[j][e];
    }
  }
}

// ---------------------------------------------------------------------------
// Host launcher. Workspace layout (bytes):
//   [0,8M)    Qb  bf16 row-major (token, feature)
//   [8M,16M)  KbT bf16 (b, h, feature, dd)  -- slab-transposed
//   [16M,24M) VbT bf16 (b, h, feature, dd)
//   [24M,+256K) mrow f32   [+256K,+512K) rzrow f32     (~24.5 MB total)
// ---------------------------------------------------------------------------
extern "C" void kernel_launch(void* const* d_in, const int* in_sizes, int n_in,
                              void* d_out, int out_size, void* d_ws, size_t ws_size,
                              hipStream_t stream) {
  const float* x    = (const float*)d_in[0];
  const float* Wq   = (const float*)d_in[1];
  const float* bq   = (const float*)d_in[2];
  const float* Wk   = (const float*)d_in[3];
  const float* bk   = (const float*)d_in[4];
  const float* Wv   = (const float*)d_in[5];
  const float* bv   = (const float*)d_in[6];
  const float* temp = (const float*)d_in[7];
  float* out = (float*)d_out;

  char* ws = (char*)d_ws;
  bf16_t* Qb   = (bf16_t*)(ws);
  bf16_t* KbT  = (bf16_t*)(ws + (size_t)8  * 1024 * 1024);
  bf16_t* VbT  = (bf16_t*)(ws + (size_t)16 * 1024 * 1024);
  float*  mrow = (float*) (ws + (size_t)24 * 1024 * 1024);
  float*  rzrw = (float*) (ws + (size_t)24 * 1024 * 1024 + 256 * 1024);

  qkv_proj<<<dim3(32, 8, 3), 256, 0, stream>>>(x, Wq, bq, Wk, bk, Wv, bv,
                                               Qb, KbT, VbT);
  attn_stats<<<dim3(64), 256, 0, stream>>>(KbT, VbT, temp, mrow, rzrw);
  attn_out<<<dim3(64, 8), 256, 0, stream>>>(Qb, KbT, VbT, temp, mrow, rzrw, out);
}
